// SAGE_14474039787718
// MI455X (gfx1250) — matrix operand order repaired and checked by reading, not compiled, not run on hardware
//
#include <hip/hip_runtime.h>
#include <hip/hip_bf16.h>
#include <math.h>

typedef __attribute__((ext_vector_type(16))) __bf16 v16bf;
typedef __attribute__((ext_vector_type(8)))  float  v8f;

#define NNODES 20000
#define KNBR   16

// ---------------------------------------------------------------------------
// fast tanh: gfx1250 has a hardware transcendental V_TANH_F32.
// ---------------------------------------------------------------------------
__device__ __forceinline__ float fast_tanh(float x) {
#if __has_builtin(__builtin_amdgcn_tanhf)
  return __builtin_amdgcn_tanhf(x);
#elif __has_builtin(__builtin_amdgcn_tanh_f32)
  return __builtin_amdgcn_tanh_f32(x);
#else
  float r;
  asm volatile("v_tanh_f32 %0, %1\n\tv_nop" : "=v"(r) : "v"(x));
  return r;
#endif
}
__device__ __forceinline__ float fast_sigmoid(float x) {
  return fmaf(0.5f, fast_tanh(0.5f * x), 0.5f);
}

// ---------------------------------------------------------------------------
// Prep: pack an f32 weight matrix into bf16 WMMA B-fragment tile order.
// B[k][n]:  trans=1 -> src[n*k_total + k]  (Wih/Whh: gates = x @ W^T)
//           trans=0 -> src[k*n_total + n]  (Wself/Wneigh: out = x @ W)
// Tile order: [nt][kc][lane(32)][16], lane l -> n = nt*16 + (l%16),
// elem e -> k = kc*32 + (l<16?0:8) + (e<8 ? e : e+8)   (CDNA5 bf16 B layout).
// ---------------------------------------------------------------------------
__global__ void pack_b_kernel(const float* __restrict__ src,
                              __bf16* __restrict__ dst,
                              int n_total, int k_total, int trans) {
  int total = n_total * k_total;
  int tid = blockIdx.x * blockDim.x + threadIdx.x;
  if (tid >= total) return;
  int KC   = k_total >> 5;
  int e    = tid & 15;
  int lane = (tid >> 4) & 31;
  int kc   = (tid >> 9) % KC;
  int nt   = tid / (512 * KC);
  int n = nt * 16 + (lane & 15);
  int k = kc * 32 + ((lane < 16) ? 0 : 8) + ((e < 8) ? e : e + 8);
  float v = trans ? src[n * k_total + k] : src[k * n_total + n];
  dst[tid] = (__bf16)v;
}

__global__ void bias_sum_kernel(const float* __restrict__ a,
                                const float* __restrict__ b,
                                float* __restrict__ o, int n) {
  int i = blockIdx.x * blockDim.x + threadIdx.x;
  if (i < n) o[i] = a[i] + b[i];
}

__global__ void f32_to_bf16_kernel(const float* __restrict__ s,
                                   __bf16* __restrict__ d, int n) {
  int i = blockIdx.x * blockDim.x + threadIdx.x;
  if (i < n) d[i] = (__bf16)s[i];
}

// ---------------------------------------------------------------------------
// Fused SAGE-LSTM layer, bf16 activations. Block = 256 threads (8 waves),
// 32 nodes per block (2 WMMA M-tiles). Each wave owns DIN/8 hidden units
// across ALL four gates -> LSTM pointwise + cell state stay in registers.
// xt and h are double-buffered in LDS: one barrier per LSTM step; the
// step-(k+1) gather overlaps step-k WMMAs; the self-feature gather is folded
// into the last step (lands in s_xt[0]).
// ---------------------------------------------------------------------------
template<int DIN, int DOUT, bool RELU, bool OUTF32>
__global__ __launch_bounds__(256)
void sage_lstm_layer(const __bf16* __restrict__ x,     // [N, DIN] bf16
                     const int*    __restrict__ nbr,   // [N, KNBR]
                     const __bf16* __restrict__ WihP,  // [4*DIN/16][DIN/32][32][16]
                     const __bf16* __restrict__ WhhP,
                     const float*  __restrict__ bg,    // bih+bhh [4*DIN]
                     const __bf16* __restrict__ WsP,   // [DOUT/16][DIN/32][32][16]
                     const __bf16* __restrict__ WnP,
                     const float*  __restrict__ bo,    // [DOUT]
                     void* __restrict__ outv)          // [N, DOUT] bf16 or f32
{
  constexpr int MT  = 32;          // nodes per block
  constexpr int KC  = DIN / 32;    // WMMA K chunks
  constexpr int HPW = DIN / 8;     // hidden units per wave
  constexpr int NTG = HPW / 16;    // N tiles per gate per wave
  constexpr int NTO = DOUT / 128;  // output N tiles per wave

  __shared__ __align__(16) __bf16 s_xt[2][MT * DIN];
  __shared__ __align__(16) __bf16 s_h [2][MT * DIN];

  const int tid   = threadIdx.x;
  const int lane  = tid & 31;
  const int wave  = tid >> 5;
  const int nbase = blockIdx.x * MT;

  // A fragment from row-major bf16 LDS (CDNA5 16-bit A layout):
  // lane<16: m=lane, K {0..7,16..23}; lane>=16: m=lane-16, K {8..15,24..31}
  auto loadA = [&](const __bf16* base, int mt, int kc) -> v16bf {
    const __bf16* p = base + (mt * 16 + (lane & 15)) * DIN
                           + kc * 32 + ((lane >= 16) ? 8 : 0);
    union { v16bf v; float4 f[2]; } u;
    u.f[0] = *(const float4*)(p);
    u.f[1] = *(const float4*)(p + 16);
    return u.v;
  };
  // B fragment: packed contiguous 16 bf16 per lane.
  auto loadB = [&](const __bf16* pk, int nt, int kc) -> v16bf {
    const __bf16* p = pk + (size_t)(((nt * KC) + kc) * 32 + lane) * 16;
    union { v16bf v; float4 f[2]; } u;
    u.f[0] = *(const float4*)(p);
    u.f[1] = *(const float4*)(p + 8);
    return u.v;
  };
  // Gather one 32-row bf16 tile into LDS (8 threads per row, b128 copies).
  // s < 0: self rows; else neighbor step s.
  auto gather = [&](__bf16* dst, int s) {
    const int node = tid >> 3, part = tid & 7;
    const int row  = (s < 0) ? (nbase + node)
                             : nbr[(size_t)(nbase + node) * KNBR + s];
    const __bf16* sp = x + (size_t)row * DIN + part * (DIN / 8);
    __bf16*       dp = dst + node * DIN + part * (DIN / 8);
    #pragma unroll
    for (int i = 0; i < DIN / 8; i += 8)
      *(float4*)(dp + i) = *(const float4*)(sp + i);
  };

  // ---- init: zero h[0], gather first neighbor step, preload gate biases ----
  {
    constexpr int per = MT * DIN / 256;
    __bf16* d = &s_h[0][tid * per];
    const float4 z = {0.f, 0.f, 0.f, 0.f};
    #pragma unroll
    for (int i = 0; i < per; i += 8) *(float4*)(d + i) = z;
  }
  gather(s_xt[0], 0);

  float gbias[4][NTG];
  #pragma unroll
  for (int g = 0; g < 4; ++g)
    #pragma unroll
    for (int nt = 0; nt < NTG; ++nt)
      gbias[g][nt] = bg[g * DIN + wave * HPW + nt * 16 + (lane & 15)];

  float cst[2][NTG][8];
  #pragma unroll
  for (int mt = 0; mt < 2; ++mt)
    #pragma unroll
    for (int nt = 0; nt < NTG; ++nt)
      #pragma unroll
      for (int r = 0; r < 8; ++r) cst[mt][nt][r] = 0.f;

  __syncthreads();

  // ---- LSTM over KNBR neighbor steps ----
  for (int step = 0; step < KNBR; ++step) {
    const int cur = step & 1;
    // Overlapped gather into the buffer nobody reads this step; on the last
    // step fetch the self features instead (ends up in s_xt[0]).
    gather(s_xt[cur ^ 1], (step + 1 < KNBR) ? (step + 1) : -1);

    v8f gacc[2][4][NTG];
    #pragma unroll
    for (int mt = 0; mt < 2; ++mt)
      #pragma unroll
      for (int g = 0; g < 4; ++g)
        #pragma unroll
        for (int nt = 0; nt < NTG; ++nt)
          #pragma unroll
          for (int r = 0; r < 8; ++r) gacc[mt][g][nt][r] = gbias[g][nt];

    #pragma unroll
    for (int g = 0; g < 4; ++g) {
      #pragma unroll
      for (int nt = 0; nt < NTG; ++nt) {
        const int ntg = (g * DIN + wave * HPW + nt * 16) >> 4;
        #pragma unroll
        for (int kc = 0; kc < KC; ++kc) {
          v16bf bi = loadB(WihP, ntg, kc);   // one B load ...
          v16bf bh = loadB(WhhP, ntg, kc);
          #pragma unroll
          for (int mt = 0; mt < 2; ++mt) {   // ... feeds both M tiles
            v16bf ax = loadA(s_xt[cur], mt, kc);
            gacc[mt][g][nt] = __builtin_amdgcn_wmma_f32_16x16x32_bf16(
                false, ax, false, bi, (short)0, gacc[mt][g][nt], false, false);
            v16bf ah = loadA(s_h[cur], mt, kc);
            gacc[mt][g][nt] = __builtin_amdgcn_wmma_f32_16x16x32_bf16(
                false, ah, false, bh, (short)0, gacc[mt][g][nt], false, false);
          }
        }
      }
    }

    // pointwise LSTM cell in registers; h -> the other LDS buffer
    #pragma unroll
    for (int mt = 0; mt < 2; ++mt) {
      #pragma unroll
      for (int nt = 0; nt < NTG; ++nt) {
        const int hidb = wave * HPW + nt * 16 + (lane & 15);
        const int mrow = mt * 16 + ((lane >= 16) ? 8 : 0);
        #pragma unroll
        for (int r = 0; r < 8; ++r) {
          float ig = fast_sigmoid(gacc[mt][0][nt][r]);
          float fg = fast_sigmoid(gacc[mt][1][nt][r]);
          float gg = fast_tanh   (gacc[mt][2][nt][r]);
          float og = fast_sigmoid(gacc[mt][3][nt][r]);
          float c2 = fmaf(fg, cst[mt][nt][r], ig * gg);
          cst[mt][nt][r] = c2;
          s_h[cur ^ 1][(mrow + r) * DIN + hidb] = (__bf16)(og * fast_tanh(c2));
        }
      }
    }
    __syncthreads();
  }

  // after step 15: final h in s_h[0], self features in s_xt[0]
  const __bf16* sxs = s_xt[0];
  const __bf16* shf = s_h[0];

  // ---- out = x @ Wself + h @ Wneigh + b (+ReLU) ----
  #pragma unroll
  for (int mt = 0; mt < 2; ++mt) {
    #pragma unroll
    for (int nt = 0; nt < NTO; ++nt) {
      const int colb = wave * (DOUT / 8) + nt * 16;
      const int ntg  = colb >> 4;
      const float bb = bo[colb + (lane & 15)];
      v8f acc;
      #pragma unroll
      for (int r = 0; r < 8; ++r) acc[r] = bb;
      #pragma unroll
      for (int kc = 0; kc < KC; ++kc) {
        v16bf b0 = loadB(WsP, ntg, kc);
        v16bf b1 = loadB(WnP, ntg, kc);
        v16bf a0 = loadA(sxs, mt, kc);
        acc = __builtin_amdgcn_wmma_f32_16x16x32_bf16(
                  false, a0, false, b0, (short)0, acc, false, false);
        v16bf a1 = loadA(shf, mt, kc);
        acc = __builtin_amdgcn_wmma_f32_16x16x32_bf16(
                  false, a1, false, b1, (short)0, acc, false, false);
      }
      const int col = colb + (lane & 15);
      const int row = nbase + mt * 16 + ((lane >= 16) ? 8 : 0);
      #pragma unroll
      for (int r = 0; r < 8; ++r) {
        float v = acc[r];
        if (RELU) v = fmaxf(v, 0.f);
        if (OUTF32)
          ((float*)outv)[(size_t)(row + r) * DOUT + col] = v;
        else
          ((__bf16*)outv)[(size_t)(row + r) * DOUT + col] = (__bf16)v;
      }
    }
  }
}

// ---------------------------------------------------------------------------
extern "C" void kernel_launch(void* const* d_in, const int* in_sizes, int n_in,
                              void* d_out, int out_size, void* d_ws, size_t ws_size,
                              hipStream_t stream) {
  (void)in_sizes; (void)n_in; (void)out_size; (void)ws_size;
  const float* xf  = (const float*)d_in[0];
  const int*   nbr = (const int*)d_in[1];

  const int dins[4]  = {128, 256, 256, 256};
  const int douts[4] = {256, 256, 256, 128};

  char* ws = (char*)d_ws;
  size_t off = 0;
  auto alloc = [&](size_t bytes) -> size_t {
    size_t o = off; off = (off + bytes + 255) & ~(size_t)255; return o;
  };

  __bf16* xbf  = (__bf16*)(ws + alloc((size_t)NNODES * 128 * 2));
  __bf16* actA = (__bf16*)(ws + alloc((size_t)NNODES * 256 * 2));
  __bf16* actB = (__bf16*)(ws + alloc((size_t)NNODES * 256 * 2));

  __bf16* wihp[4]; __bf16* whhp[4]; __bf16* wsp[4]; __bf16* wnp[4]; float* bgp[4];
  for (int l = 0; l < 4; ++l) {
    int din = dins[l], dout = douts[l];
    wihp[l] = (__bf16*)(ws + alloc((size_t)4 * din * din * 2));
    whhp[l] = (__bf16*)(ws + alloc((size_t)4 * din * din * 2));
    wsp[l]  = (__bf16*)(ws + alloc((size_t)din * dout * 2));
    wnp[l]  = (__bf16*)(ws + alloc((size_t)din * dout * 2));
    bgp[l]  = (float*)(ws + alloc((size_t)4 * din * 4));
  }

  // ---- prep: repack weights, fuse gate biases, convert layer-1 input ----
  f32_to_bf16_kernel<<<(NNODES * 128 + 255) / 256, 256, 0, stream>>>(
      xf, xbf, NNODES * 128);
  for (int l = 0; l < 4; ++l) {
    int din = dins[l], dout = douts[l];
    const float* Wih = (const float*)d_in[2 + 7 * l + 0];
    const float* Whh = (const float*)d_in[2 + 7 * l + 1];
    const float* bih = (const float*)d_in[2 + 7 * l + 2];
    const float* bhh = (const float*)d_in[2 + 7 * l + 3];
    const float* Wsf = (const float*)d_in[2 + 7 * l + 4];
    const float* Wng = (const float*)d_in[2 + 7 * l + 5];

    int tg = 4 * din * din;
    pack_b_kernel<<<(tg + 255) / 256, 256, 0, stream>>>(Wih, wihp[l], 4 * din, din, 1);
    pack_b_kernel<<<(tg + 255) / 256, 256, 0, stream>>>(Whh, whhp[l], 4 * din, din, 1);
    int to = din * dout;
    pack_b_kernel<<<(to + 255) / 256, 256, 0, stream>>>(Wsf, wsp[l], dout, din, 0);
    pack_b_kernel<<<(to + 255) / 256, 256, 0, stream>>>(Wng, wnp[l], dout, din, 0);
    bias_sum_kernel<<<(4 * din + 255) / 256, 256, 0, stream>>>(bih, bhh, bgp[l], 4 * din);
  }

  const int blocks = NNODES / 32;  // 625

  const float* b1 = (const float*)d_in[2 + 0 * 7 + 6];
  const float* b2 = (const float*)d_in[2 + 1 * 7 + 6];
  const float* b3 = (const float*)d_in[2 + 2 * 7 + 6];
  const float* b4 = (const float*)d_in[2 + 3 * 7 + 6];

  sage_lstm_layer<128, 256, true,  false><<<blocks, 256, 0, stream>>>(
      xbf,  nbr, wihp[0], whhp[0], bgp[0], wsp[0], wnp[0], b1, actA);
  sage_lstm_layer<256, 256, true,  false><<<blocks, 256, 0, stream>>>(
      actA, nbr, wihp[1], whhp[1], bgp[1], wsp[1], wnp[1], b2, actB);
  sage_lstm_layer<256, 256, true,  false><<<blocks, 256, 0, stream>>>(
      actB, nbr, wihp[2], whhp[2], bgp[2], wsp[2], wnp[2], b3, actA);
  sage_lstm_layer<256, 128, false, true ><<<blocks, 256, 0, stream>>>(
      actA, nbr, wihp[3], whhp[3], bgp[3], wsp[3], wnp[3], b4, (float*)d_out);
}
